// CausalMHA_67843303408009
// MI455X (gfx1250) — matrix operand order repaired and checked by reading, not compiled, not run on hardware
//
#include <hip/hip_runtime.h>
#include <hip/hip_bf16.h>
#include <stdint.h>

// ---------------------------------------------------------------------------
// Causal MHA forward on MI455X (gfx1250).
//  - All GEMM stages via v_wmma_f32_16x16x32_bf16 (wave32 WMMA).
//  - Weight tiles staged into LDS with the Tensor Data Mover
//    (tensor_load_to_lds + s_wait_tensorcnt), double buffered, shared by
//    the 8 waves of a workgroup (8x B-operand reuse).
//  - B-tile read back through an explicit address_space(3) pointer so the
//    compiler emits ds_load_b128 (not flat) and cannot fold the TDM-written
//    data to undef.
// ---------------------------------------------------------------------------

typedef __bf16 bf16_t;
typedef __attribute__((ext_vector_type(16))) __bf16 v16bf;
typedef __attribute__((ext_vector_type(8)))  __bf16 v8bf;
typedef __attribute__((ext_vector_type(8)))  float  v8f;
typedef __attribute__((ext_vector_type(4)))  uint32_t v4u;
typedef __attribute__((ext_vector_type(8)))  int      v8i;
typedef __attribute__((ext_vector_type(4)))  int      v4i;

#define LDS_AS __attribute__((address_space(3)))

#define MHA_B  2
#define MHA_T  2048
#define MHA_D  1024
#define MHA_H  16
#define MHA_HD 64
#define MHA_M  (MHA_B * MHA_T)     // 4096 activation rows
#define MHA_3D (3 * MHA_D)         // 3072

__device__ __forceinline__ v8f wmma_bf16(v16bf a, v16bf b, v8f c) {
  return __builtin_amdgcn_wmma_f32_16x16x32_bf16(
      false, a, false, b, (short)0, c, false, false);
}

__device__ __forceinline__ v16bf frag_pair(const bf16_t* p0, const bf16_t* p1) {
  union { v16bf v; v8bf h[2]; } u;
  u.h[0] = *(const v8bf*)p0;
  u.h[1] = *(const v8bf*)p1;
  return u.v;
}

__device__ __forceinline__ v16bf frag_pair_lds(const LDS_AS bf16_t* p0,
                                               const LDS_AS bf16_t* p1) {
  union { v16bf v; v8bf h[2]; } u;
  u.h[0] = *(const LDS_AS v8bf*)p0;
  u.h[1] = *(const LDS_AS v8bf*)p1;
  return u.v;
}

// A-matrix (16x32) lane layout: row m = lane&15; K = k0 + half*16 + [0..15].
__device__ __forceinline__ v16bf load_a_frag(const bf16_t* row, int k0, int half) {
  return frag_pair(row + k0 + half * 16, row + k0 + half * 16 + 8);
}

// B-matrix (32x16) from column-major storage (column n contiguous in K):
// lane holds column n = lane&15, K = k0 + half*16 + [0..15].
__device__ __forceinline__ v16bf load_b_frag_col(const bf16_t* col, int k0, int half) {
  return frag_pair(col + k0 + half * 16, col + k0 + half * 16 + 8);
}

// ---------------------------------------------------------------------------
// Tensor Data Mover: stage a 2D weight tile (64 rows x 32 K-elems, bf16)
// into LDS. Descriptor built per CDNA5 ISA D# group0/group1 layout.
// ---------------------------------------------------------------------------
#if __has_builtin(__builtin_amdgcn_tensor_load_to_lds)
#define MHA_HAVE_TDM 1
#else
#define MHA_HAVE_TDM 0
#endif

__device__ __forceinline__ void tdm_load_b_tile(uint32_t lds_off_bytes,
                                                const bf16_t* gsrc_tile,
                                                uint32_t tensor_d0,   // K elems
                                                uint32_t tensor_d1,   // rows
                                                uint32_t stride0) {   // elems
#if MHA_HAVE_TDM
  const uint64_t ga = (uint64_t)(uintptr_t)gsrc_tile;
  v4u g0;
  g0.x = 1u;                                         // count=1, user descriptor
  g0.y = lds_off_bytes;                              // LDS byte address
  g0.z = (uint32_t)ga;                               // global addr [95:64]
  g0.w = (uint32_t)((ga >> 32) & 0x01FFFFFFu)        // global addr [120:96]
         | 0x80000000u;                              // type=2 @ [127:126]
  v8i g1;
  g1[0] = (int)(1u << 16);                           // data_size=1 (2 bytes)
  g1[1] = (int)((tensor_d0 & 0xFFFFu) << 16);        // tensor_dim0 lo16 @ [63:48]
  g1[2] = (int)((tensor_d0 >> 16) | ((tensor_d1 & 0xFFFFu) << 16));
  g1[3] = (int)((tensor_d1 >> 16) | (32u << 16));    // tile_dim0 = 32
  g1[4] = (int)64u;                                  // tile_dim1 = 64
  g1[5] = (int)stride0;                              // dim0 stride lo32
  g1[6] = 0;
  g1[7] = 0;
  v4i g2 = (v4i)0;
  v4i g3 = (v4i)0;
#if defined(__clang_major__) && (__clang_major__ >= 23)
  v8i g4 = (v8i)0;
  __builtin_amdgcn_tensor_load_to_lds(g0, g1, g2, g3, g4, 0);
#else
  __builtin_amdgcn_tensor_load_to_lds(g0, g1, g2, g3, 0);
#endif
#else
  (void)lds_off_bytes; (void)gsrc_tile; (void)tensor_d0; (void)tensor_d1;
  (void)stride0;
#endif
}

__device__ __forceinline__ void tdm_wait(int pending) {
#if MHA_HAVE_TDM
  if (pending) __builtin_amdgcn_s_wait_tensorcnt((short)1);
  else         __builtin_amdgcn_s_wait_tensorcnt((short)0);
#else
  (void)pending;
#endif
}

// ---------------------------------------------------------------------------
// Kernel 0: f32 -> bf16 conversion (grid-stride)
// ---------------------------------------------------------------------------
__global__ void mha_cvt_bf16(const float* __restrict__ src,
                             bf16_t* __restrict__ dst, int n) {
  for (int i = blockIdx.x * blockDim.x + threadIdx.x; i < n;
       i += blockDim.x * gridDim.x) {
    dst[i] = (bf16_t)src[i];
  }
}

// ---------------------------------------------------------------------------
// Shared-B WMMA GEMM core: block = 8 waves = 128 rows x 64 cols of output.
// B tile (64 rows x 32 K) staged in LDS via TDM, double buffered.
// ---------------------------------------------------------------------------
__device__ __forceinline__ void gemm_core_128x64(
    const bf16_t* __restrict__ arow,   // this lane's A row pointer
    const bf16_t* __restrict__ wb,     // weights [rows, K] row-major
    int n0, int rows_total, int K,
    bf16_t* btile,                     // __shared__ [2][64*32], LDS offset 0
    v8f acc[4]) {
  const int lane = threadIdx.x & 31;
  const int half = lane >> 4;
  const int nlo  = lane & 15;
  const bool issuer = (threadIdx.x >> 5) == 0;   // wave 0 drives the TDM
  const int nsteps = K / 32;

#if MHA_HAVE_TDM
  // Pin the LDS allocation (TDM writes it behind the compiler's back).
  {
    bf16_t* keep = btile;
    asm volatile("" : : "v"(keep));
  }
  if (issuer)
    tdm_load_b_tile(0u, wb + (size_t)n0 * K, (uint32_t)K,
                    (uint32_t)rows_total, (uint32_t)K);
#endif

  for (int s = 0; s < nsteps; ++s) {
    const int cur = s & 1;
    const int k0  = s * 32;
#if MHA_HAVE_TDM
    __syncthreads();                       // buf[cur^1] consumers are done
    if (s + 1 < nsteps && issuer)
      tdm_load_b_tile((uint32_t)((cur ^ 1) * 64 * 32 * 2),
                      wb + (size_t)n0 * K + (k0 + 32), (uint32_t)K,
                      (uint32_t)rows_total, (uint32_t)K);
    if (issuer) tdm_wait(s + 1 < nsteps);  // buf[cur] is complete
    __syncthreads();                       // publish buf[cur]

    __builtin_prefetch(arow + k0 + 128, 0, 1);     // global_prefetch_b8

    // AS(3) pointer built from the integer LDS offset (btile lives at LDS 0,
    // matching the TDM descriptor). inttoptr provenance keeps the ds_loads
    // alive and ordered; launder for extra safety.
    LDS_AS bf16_t* bt = (LDS_AS bf16_t*)(uint32_t)(cur * (64 * 32 * 2));
    asm volatile("" : "+v"(bt) : : "memory");

    v16bf a = load_a_frag(arow, k0, half);
#pragma unroll
    for (int j = 0; j < 4; ++j) {
      const LDS_AS bf16_t* bcol = bt + (j * 16 + nlo) * 32;
      acc[j] = wmma_bf16(
          a, frag_pair_lds(bcol + half * 16, bcol + half * 16 + 8), acc[j]);
    }
#else
    __syncthreads();
    {   // cooperative copy fallback: 256 threads x 16B
      const int idx = threadIdx.x;
      const int r = idx >> 2;
      const int c = (idx & 3) * 8;
      *(v8bf*)&btile[cur * (64 * 32) + r * 32 + c] =
          *(const v8bf*)(wb + (size_t)(n0 + r) * K + k0 + c);
    }
    __syncthreads();
    __builtin_prefetch(arow + k0 + 128, 0, 1);
    const bf16_t* bt = btile + cur * (64 * 32);
    v16bf a = load_a_frag(arow, k0, half);
#pragma unroll
    for (int j = 0; j < 4; ++j) {
      const bf16_t* bcol = bt + (j * 16 + nlo) * 32;
      acc[j] = wmma_bf16(a, load_b_frag_col(bcol, 0, half), acc[j]);
    }
#endif
  }
}

// ---------------------------------------------------------------------------
// Kernel 1: QKV GEMM.  out[m,n] = sum_k x[m,k]*Wqkv[n,k] + bqkv[n]
// Q,K stored [B,H,T,HD]; V stored transposed [B,H,HD,T].
// ---------------------------------------------------------------------------
__global__ void __launch_bounds__(256)
mha_qkv_gemm(const bf16_t* __restrict__ xb, const bf16_t* __restrict__ wb,
             const float* __restrict__ bias, bf16_t* __restrict__ qh,
             bf16_t* __restrict__ kh, bf16_t* __restrict__ vT) {
  __shared__ bf16_t btile[2 * 64 * 32];    // only LDS object -> offset 0

  const int lane = threadIdx.x & 31;
  const int half = lane >> 4;
  const int nlo  = lane & 15;
  const int wave = threadIdx.x >> 5;

  const int NT = MHA_3D / 64;              // 48 column tiles
  const int bm = blockIdx.x / NT;          // 32 row blocktiles (128 rows each)
  const int nt = blockIdx.x - bm * NT;
  const int m0 = bm * 128 + wave * 16;
  const int n0 = nt * 64;

  const bf16_t* arow = xb + (size_t)(m0 + nlo) * MHA_D;

  v8f acc[4];
#pragma unroll
  for (int j = 0; j < 4; ++j) acc[j] = (v8f)0.0f;

  gemm_core_128x64(arow, wb, n0, MHA_3D, MHA_D, btile, acc);

#pragma unroll
  for (int j = 0; j < 4; ++j) {
#pragma unroll
    for (int v = 0; v < 8; ++v) {
      const int row = m0 + v + half * 8;        // flattened (b, t)
      const int n   = n0 + j * 16 + nlo;        // 0..3071
      const float val = acc[j][v] + bias[n];
      const int which = n >> 10;                // 0=q,1=k,2=v
      const int d     = n & (MHA_D - 1);
      const int h     = d >> 6;
      const int hd    = d & (MHA_HD - 1);
      const int bb    = row >> 11;
      const int t     = row & (MHA_T - 1);
      if (which == 0) {
        qh[(((size_t)bb * MHA_H + h) * MHA_T + t) * MHA_HD + hd] = (bf16_t)val;
      } else if (which == 1) {
        kh[(((size_t)bb * MHA_H + h) * MHA_T + t) * MHA_HD + hd] = (bf16_t)val;
      } else {
        vT[(((size_t)bb * MHA_H + h) * MHA_HD + hd) * MHA_T + t] = (bf16_t)val;
      }
    }
  }
}

// ---------------------------------------------------------------------------
// Kernel 2: causal flash attention. One wave owns 16 query rows of one
// (b,h). Per 32-key block: 4 WMMAs for S=QK^T, online softmax with 16-lane
// shuffles, P restaged through per-wave LDS, 4 WMMAs for P*V.
// ---------------------------------------------------------------------------
__global__ void __launch_bounds__(256)
mha_flash(const bf16_t* __restrict__ qh, const bf16_t* __restrict__ kh,
          const bf16_t* __restrict__ vT, bf16_t* __restrict__ ctx) {
  __shared__ bf16_t plds[8 * 16 * 32];

  const int lane  = threadIdx.x & 31;
  const int half  = lane >> 4;
  const int nlo   = lane & 15;
  const int wslot = threadIdx.x >> 5;
  const int wid   = blockIdx.x * (blockDim.x >> 5) + wslot;

  const int QT = MHA_T / 16;
  const int qt = wid & (QT - 1);
  const int h  = (wid >> 7) & (MHA_H - 1);
  const int bb = wid >> 11;
  if (bb >= MHA_B) return;
  const int q0 = qt * 16;

  const bf16_t* Qbh = qh + ((size_t)bb * MHA_H + h) * MHA_T * MHA_HD;
  const bf16_t* Kbh = kh + ((size_t)bb * MHA_H + h) * MHA_T * MHA_HD;
  const bf16_t* Vbh = vT + ((size_t)bb * MHA_H + h) * MHA_HD * MHA_T;

  const bf16_t* qrow = Qbh + (size_t)(q0 + nlo) * MHA_HD;
  const v16bf qa0 = load_a_frag(qrow, 0, half);
  const v16bf qa1 = load_a_frag(qrow, 32, half);

  float rmax[8], rsum[8];
  v8f oacc[4];
#pragma unroll
  for (int v = 0; v < 8; ++v) { rmax[v] = -3.0e38f; rsum[v] = 0.0f; }
#pragma unroll
  for (int j = 0; j < 4; ++j) oacc[j] = (v8f)0.0f;

  bf16_t* plw = plds + wslot * (16 * 32);
  const float scale = 0.125f;
  const int kb_end = (q0 + 15) >> 5;

  for (int kb = 0; kb <= kb_end; ++kb) {
    const int k0b = kb * 32;

    v8f s[2];
#pragma unroll
    for (int sub = 0; sub < 2; ++sub) {
      const bf16_t* kcol = Kbh + (size_t)(k0b + sub * 16 + nlo) * MHA_HD;
      v8f a = (v8f)0.0f;
      a = wmma_bf16(qa0, load_b_frag_col(kcol, 0, half), a);
      a = wmma_bf16(qa1, load_b_frag_col(kcol, 32, half), a);
      s[sub] = a;
    }

#pragma unroll
    for (int sub = 0; sub < 2; ++sub) {
      const int kg = k0b + sub * 16 + nlo;
#pragma unroll
      for (int v = 0; v < 8; ++v) {
        const int qg = q0 + v + half * 8;
        float val = s[sub][v] * scale;
        s[sub][v] = (kg > qg) ? -3.0e38f : val;
      }
    }

    float corr[8];
#pragma unroll
    for (int v = 0; v < 8; ++v) {
      float a0 = s[0][v], a1 = s[1][v];
      float mx = fmaxf(a0, a1);
#pragma unroll
      for (int off = 1; off < 16; off <<= 1)
        mx = fmaxf(mx, __shfl_xor(mx, off, 32));
      const float nm = fmaxf(rmax[v], mx);
      const float c  = __expf(rmax[v] - nm);
      const float p0 = __expf(a0 - nm);
      const float p1 = __expf(a1 - nm);
      float ps = p0 + p1;
#pragma unroll
      for (int off = 1; off < 16; off <<= 1)
        ps += __shfl_xor(ps, off, 32);
      rsum[v] = rsum[v] * c + ps;
      rmax[v] = nm;
      corr[v] = c;
      const int m = v + half * 8;
      plw[m * 32 + nlo]      = (bf16_t)p0;
      plw[m * 32 + 16 + nlo] = (bf16_t)p1;
    }

#pragma unroll
    for (int j = 0; j < 4; ++j)
#pragma unroll
      for (int v = 0; v < 8; ++v) oacc[j][v] *= corr[v];

    const bf16_t* prow = plw + nlo * 32;
    const v16bf pa = load_a_frag(prow, 0, half);
#pragma unroll
    for (int j = 0; j < 4; ++j) {
      const bf16_t* vcol = Vbh + (size_t)(j * 16 + nlo) * MHA_T + k0b;
      oacc[j] = wmma_bf16(pa, frag_pair(vcol + half * 16, vcol + half * 16 + 8),
                          oacc[j]);
    }
  }

#pragma unroll
  for (int v = 0; v < 8; ++v) rsum[v] = 1.0f / rsum[v];
#pragma unroll
  for (int j = 0; j < 4; ++j) {
#pragma unroll
    for (int v = 0; v < 8; ++v) {
      const int t = q0 + v + half * 8;
      const int n = h * MHA_HD + j * 16 + nlo;
      ctx[((size_t)bb * MHA_T + t) * MHA_D + n] = (bf16_t)(oacc[j][v] * rsum[v]);
    }
  }
}

// ---------------------------------------------------------------------------
// Kernel 3: output projection. out[m,n] = sum_k ctx[m,k]*Wproj[n,k]+bproj[n]
// ---------------------------------------------------------------------------
__global__ void __launch_bounds__(256)
mha_proj_gemm(const bf16_t* __restrict__ cb, const bf16_t* __restrict__ wb,
              const float* __restrict__ bias, float* __restrict__ out) {
  __shared__ bf16_t btile[2 * 64 * 32];    // only LDS object -> offset 0

  const int lane = threadIdx.x & 31;
  const int half = lane >> 4;
  const int nlo  = lane & 15;
  const int wave = threadIdx.x >> 5;

  const int NT = MHA_D / 64;               // 16 column tiles
  const int bm = blockIdx.x / NT;          // 32 row blocktiles
  const int nt = blockIdx.x - bm * NT;
  const int m0 = bm * 128 + wave * 16;
  const int n0 = nt * 64;

  const bf16_t* arow = cb + (size_t)(m0 + nlo) * MHA_D;

  v8f acc[4];
#pragma unroll
  for (int j = 0; j < 4; ++j) acc[j] = (v8f)0.0f;

  gemm_core_128x64(arow, wb, n0, MHA_D, MHA_D, btile, acc);

#pragma unroll
  for (int j = 0; j < 4; ++j) {
#pragma unroll
    for (int v = 0; v < 8; ++v) {
      const int row = m0 + v + half * 8;
      const int n   = n0 + j * 16 + nlo;
      out[(size_t)row * MHA_D + n] = acc[j][v] + bias[n];
    }
  }
}

// ---------------------------------------------------------------------------
// Launch
// ---------------------------------------------------------------------------
extern "C" void kernel_launch(void* const* d_in, const int* in_sizes, int n_in,
                              void* d_out, int out_size, void* d_ws, size_t ws_size,
                              hipStream_t stream) {
  (void)in_sizes; (void)n_in; (void)out_size; (void)ws_size;

  const float* x     = (const float*)d_in[0];
  const float* Wqkv  = (const float*)d_in[1];
  const float* bqkv  = (const float*)d_in[2];
  const float* Wproj = (const float*)d_in[3];
  const float* bproj = (const float*)d_in[4];
  float* out = (float*)d_out;

  char* ws = (char*)d_ws;
  size_t off = 0;
  bf16_t* xb     = (bf16_t*)(ws + off); off += (size_t)MHA_M  * MHA_D * 2;
  bf16_t* wqkvb  = (bf16_t*)(ws + off); off += (size_t)MHA_3D * MHA_D * 2;
  bf16_t* wprojb = (bf16_t*)(ws + off); off += (size_t)MHA_D  * MHA_D * 2;
  bf16_t* qh     = (bf16_t*)(ws + off); off += (size_t)MHA_M  * MHA_D * 2;
  bf16_t* kh     = (bf16_t*)(ws + off); off += (size_t)MHA_M  * MHA_D * 2;
  bf16_t* vT     = (bf16_t*)(ws + off); off += (size_t)MHA_M  * MHA_D * 2;
  bf16_t* ctx    = (bf16_t*)(ws + off); off += (size_t)MHA_M  * MHA_D * 2;

  mha_cvt_bf16<<<2048, 256, 0, stream>>>(x, xb, MHA_M * MHA_D);
  mha_cvt_bf16<<<2048, 256, 0, stream>>>(Wqkv, wqkvb, MHA_3D * MHA_D);
  mha_cvt_bf16<<<1024, 256, 0, stream>>>(Wproj, wprojb, MHA_D * MHA_D);

  // QKV: (4096/128 row blocks) * (3072/64 col tiles) = 1536 blocks
  mha_qkv_gemm<<<1536, 256, 0, stream>>>(xb, wqkvb, bqkv, qh, kh, vT);

  // flash attention: B*H*(T/16) = 4096 waves, 8 per block
  mha_flash<<<512, 256, 0, stream>>>(qh, kh, vT, ctx);

  // Proj: (4096/128) * (1024/64) = 512 blocks
  mha_proj_gemm<<<512, 256, 0, stream>>>(ctx, wprojb, bproj, out);
}
